// DNCS2S_81217831567538
// MI455X (gfx1250) — compile-verified, matrix operand order checked by reading, not run on hardware
//
#include <hip/hip_runtime.h>
#include <hip/hip_bf16.h>
#include <math.h>

// ---------------- problem constants ----------------
#define B_    32
#define S_    64
#define T_    64
#define D_    512
#define R_    4
#define M_    64
#define V_    10000
#define VP_   10048         // V padded to multiple of 64 (WMMA tile group)
#define IF_   3607          // R*D + 3*D + 5*R + 3
#define IFP_  3616          // IF padded to multiple of 16 (row stride of xi)
#define IFP64_ 3648         // IF padded to multiple of 64 (Wi weight rows)
#define CLIP_ 20.0f
#define EPS_  1e-6f

typedef __attribute__((ext_vector_type(16))) _Float16 v16h;
typedef __attribute__((ext_vector_type(8)))  _Float16 v8h;
typedef __attribute__((ext_vector_type(8)))  float    v8f;

__device__ __forceinline__ float sigm(float x){ return 1.0f/(1.0f+expf(-x)); }
__device__ __forceinline__ float softplus_(float x){ return (x>20.0f)? x : log1pf(expf(x)); }
__device__ __forceinline__ float wave_sum(float s){
  #pragma unroll
  for (int off=16; off; off>>=1) s += __shfl_xor(s, off, 32);
  return s;
}

// ======================================================================
// WMMA GEMM:  C[M,N] = A[M,K](f32) * W[Npad,K](f16)^T  (+bias) (+=C)
// One wave: 16-row M tile x 4 N tiles of 16.  wave32, v_wmma_f32_16x16x32_f16.
// REQUIREMENT: Npad % 64 == 0  ->  all 4 tiles always valid, branchless k-loop.
// Per k-step: stage all 4 B fragments in distinct regs first (one load clause,
// descending loadcnt waits), then convert A, then 4 back-to-back WMMAs.
// A-fragment layout (16-bit A 16x32): lane l holds row m = l&15,
//   half e -> K = k0 + 8*(l>=16) + e + 8*(e>=8)  => two contiguous 8-half runs.
// B-fragment is the transpose-role layout; a B "column" n is W row n, so the
//   same two 16-byte contiguous loads apply.
// TRANS_STORE==1: store C[m,n] -> out[(m/64)*V*T + n*T + (m%64)]  (logits^T)
// ======================================================================
template<int TRANS_STORE>
__global__ void k_wmma_gemm(const float* __restrict__ A, int lda,
                            const _Float16* __restrict__ W, int ldw,
                            const float* __restrict__ bias,
                            float* __restrict__ C, int ldc,
                            int Mrows, int N, int Npad, int K, int accum)
{
  const int lane = threadIdx.x & 31;
  const int wid  = (int)((blockIdx.x * blockDim.x + threadIdx.x) >> 5);
  const int mtiles = Mrows >> 4;
  const int mt = wid % mtiles;
  const int ng = wid / mtiles;
  const int n0base = ng * 64;
  if (n0base >= Npad) return;                 // wave-uniform (grid overshoot only)
  const int row   = lane & 15;
  const int khalf = (lane & 16) ? 8 : 0;
  const int m0    = mt << 4;

  v8f acc[4] = {};
  const float* arow = A + (size_t)(m0 + row) * (size_t)lda + khalf;
  const _Float16* wr[4];
  #pragma unroll
  for (int t = 0; t < 4; ++t)
    wr[t] = W + (size_t)(n0base + t * 16 + row) * (size_t)ldw + khalf;

  for (int k0 = 0; k0 < K; k0 += 32) {
    // stage all B fragments (8 x b128 loads in one clause)
    v8h blo[4], bhi[4];
    #pragma unroll
    for (int t = 0; t < 4; ++t) {
      const v8h* bp = (const v8h*)(wr[t] + k0);
      blo[t] = bp[0];                         // K = kb .. kb+7
      bhi[t] = bp[2];                         // K = kb+16 .. kb+23
    }
    // A fragment: load f32, packed-convert to f16
    v16h a;
    const float* ap = arow + k0;
    #pragma unroll
    for (int e = 0; e < 8; ++e) {
      a[e]     = (_Float16)ap[e];
      a[e + 8] = (_Float16)ap[e + 16];
    }
    // 4 WMMAs, accumulators persistent
    #pragma unroll
    for (int t = 0; t < 4; ++t) {
      v16h bfr;
      #pragma unroll
      for (int e = 0; e < 8; ++e) { bfr[e] = blo[t][e]; bfr[e + 8] = bhi[t][e]; }
      acc[t] = __builtin_amdgcn_wmma_f32_16x16x32_f16(
          false, a, false, bfr, (short)0, acc[t], false, false);
    }
  }

  const int mbase = m0 + ((lane & 16) ? 8 : 0);
  const int ncol  = lane & 15;
  #pragma unroll
  for (int t = 0; t < 4; ++t) {
    int n = n0base + t * 16 + ncol;
    if (n >= N) continue;                     // store-only guard (zero-padded W rows)
    float bv = bias ? bias[n] : 0.0f;
    #pragma unroll
    for (int r = 0; r < 8; ++r) {
      int m = mbase + r;
      float v = acc[t][r] + bv;
      if (TRANS_STORE) {
        int bb = m >> 6, tt = m & 63;
        C[((size_t)bb * V_ + n) * T_ + tt] = v;
      } else {
        size_t o = (size_t)m * (size_t)ldc + n;
        C[o] = accum ? (C[o] + v) : v;
      }
    }
  }
}

// ---------------- utility / elementwise kernels ----------------
__global__ void k_zero(float* p, long n){
  long i = (long)blockIdx.x * blockDim.x + threadIdx.x;
  if (i < n) p[i] = 0.0f;
}

// f32 [N,ldsrc] -> f16 [Npad,Kd], rows >= N zero-filled
__global__ void k_convert(const float* __restrict__ src, _Float16* __restrict__ dst,
                          int N, int ldsrc, int Kd, int total){
  int idx = blockIdx.x * blockDim.x + threadIdx.x;
  if (idx >= total) return;
  int n = idx / Kd, k = idx - n * Kd;
  dst[idx] = (n < N) ? (_Float16)src[(size_t)n * ldsrc + k] : (_Float16)0.0f;
}

__global__ void k_copy(float* __restrict__ d, const float* __restrict__ s, int n){
  int i = blockIdx.x * blockDim.x + threadIdx.x;
  if (i < n) d[i] = s[i];
}

__global__ void k_embed(const float* __restrict__ emb, const int* __restrict__ ids,
                        int step, int slen, float* __restrict__ x){
  int i = blockIdx.x * blockDim.x + threadIdx.x;
  if (i >= B_ * D_) return;
  int b = i >> 9, d = i & 511;
  x[i] = emb[(size_t)ids[b * slen + step] * D_ + d];
}

// g[32,2048] (i,f,g,o chunks of 512) ; h,c in-place update
__global__ void k_lstm(const float* __restrict__ g, float* __restrict__ h, float* __restrict__ c){
  int i = blockIdx.x * blockDim.x + threadIdx.x;
  if (i >= B_ * D_) return;
  int b = i >> 9, d = i & 511;
  const float* gb = g + (size_t)b * 2048;
  float gi = gb[d], gf = gb[512 + d], gg = gb[1024 + d], go = gb[1536 + d];
  float cn = sigm(gf) * c[i] + sigm(gi) * tanhf(gg);
  c[i] = cn;
  h[i] = sigm(go) * tanhf(cn);
}

__global__ void k_clip(const float* __restrict__ h, float* __restrict__ o){
  int i = blockIdx.x * blockDim.x + threadIdx.x;
  if (i >= B_ * D_) return;
  o[i] = fminf(fmaxf(h[i], -CLIP_), CLIP_);
}

// xi activations -> xia  (tanh / softplus / sigmoid per region; modes kept raw)
__global__ void k_xiact(const float* __restrict__ xi, float* __restrict__ xia){
  int idx = blockIdx.x * blockDim.x + threadIdx.x;
  if (idx >= B_ * IF_) return;
  int b = idx / IF_, j = idx - b * IF_;
  float x = xi[(size_t)b * IFP_ + j], y;
  if      (j < 2048) y = tanhf(x);        // rk
  else if (j < 2052) y = softplus_(x);    // rs
  else if (j < 2564) y = tanhf(x);        // wk
  else if (j == 2564) y = softplus_(x);   // ws
  else if (j < 3077) y = sigm(x);         // er
  else if (j < 3589) y = tanhf(x);        // wv
  else if (j < 3595) y = sigm(x);         // fg, ag, wg
  else               y = x;               // modes logits
  xia[(size_t)b * IFP_ + j] = y;
}

__global__ void k_modes(const float* __restrict__ xia, float* __restrict__ modes){
  int idx = threadIdx.x;
  if (idx >= B_ * R_) return;
  int b = idx >> 2, r = idx & 3;
  const float* p = xia + (size_t)b * IFP_ + 3595 + r * 3;
  float a = p[0], c = p[1], d = p[2];
  float mx = fmaxf(a, fmaxf(c, d));
  float ea = expf(a - mx), eb = expf(c - mx), ec = expf(d - mx);
  float s = ea + eb + ec;
  float* o = modes + b * 12 + r * 3;
  o[0] = ea / s; o[1] = eb / s; o[2] = ec / s;
}

// usage = (usage + (1-usage)*ww_prev) * prod_r(1 - fg_r * rw_prev[r,m])
__global__ void k_usage(const float* __restrict__ xia, const float* __restrict__ ww,
                        const float* __restrict__ rw, float* __restrict__ usage){
  int idx = blockIdx.x * blockDim.x + threadIdx.x;
  if (idx >= B_ * M_) return;
  int b = idx >> 6, m = idx & 63;
  float u = usage[idx];
  u = u + (1.0f - u) * ww[idx];
  const float* xb = xia + (size_t)b * IFP_;
  float pr = 1.0f;
  #pragma unroll
  for (int r = 0; r < R_; ++r) pr *= (1.0f - xb[3589 + r] * rw[b * (R_*M_) + r * M_ + m]);
  usage[idx] = u * pr;
}

// generic wave-per-row L2 norm of 512-vectors: base = (i/rpb)*os + (i%rpb)*is + off
__global__ void k_norm512(const float* __restrict__ src, float* __restrict__ out,
                          int count, int rpb, long os, long is, long off){
  int wid = (int)((blockIdx.x * blockDim.x + threadIdx.x) >> 5);
  int lane = threadIdx.x & 31;
  if (wid >= count) return;
  long base = (long)(wid / rpb) * os + (long)(wid % rpb) * is + off;
  float s = 0.0f;
  for (int j = lane; j < D_; j += 32) { float v = src[base + j]; s += v * v; }
  s = wave_sum(s);
  if (lane == 0) out[wid] = sqrtf(s);
}

// write-key content scores (pre-softmax), wave per (b,m)
__global__ void k_wcw(const float* __restrict__ mem, const float* __restrict__ xia,
                      const float* __restrict__ mnorm, const float* __restrict__ wknorm,
                      float* __restrict__ wcw){
  int wid = (int)((blockIdx.x * blockDim.x + threadIdx.x) >> 5);
  int lane = threadIdx.x & 31;
  if (wid >= B_ * M_) return;
  int b = wid >> 6;
  const float* mv = mem + (size_t)wid * D_;
  const float* kv = xia + (size_t)b * IFP_ + 2052;
  float s = 0.0f;
  for (int j = lane; j < D_; j += 32) s += mv[j] * kv[j];
  s = wave_sum(s);
  if (lane == 0) {
    float wsv = xia[(size_t)b * IFP_ + 2564];
    wcw[wid] = s / ((mnorm[wid] + EPS_) * (wknorm[b] + EPS_)) * wsv;
  }
}

// read-key scores, wave per (b,r,m)
__global__ void k_cw(const float* __restrict__ mem, const float* __restrict__ xia,
                     const float* __restrict__ mnorm, const float* __restrict__ rknorm,
                     float* __restrict__ cw){
  int wid = (int)((blockIdx.x * blockDim.x + threadIdx.x) >> 5);
  int lane = threadIdx.x & 31;
  if (wid >= B_ * R_ * M_) return;
  int b = wid >> 8, r = (wid >> 6) & 3, m = wid & 63;
  const float* mv = mem + ((size_t)b * M_ + m) * D_;
  const float* kv = xia + (size_t)b * IFP_ + r * D_;
  float s = 0.0f;
  for (int j = lane; j < D_; j += 32) s += mv[j] * kv[j];
  s = wave_sum(s);
  if (lane == 0) {
    float rs = xia[(size_t)b * IFP_ + 2048 + r];
    cw[wid] = s / ((mnorm[b * M_ + m] + EPS_) * (rknorm[b * R_ + r] + EPS_)) * rs;
  }
}

// softmax over last dim 64; one block (64 threads) per row; in-place safe
__global__ void k_softmax64(const float* __restrict__ src, float* __restrict__ dst){
  __shared__ float s[64];
  int row = blockIdx.x, t = threadIdx.x;
  float v = src[row * 64 + t];
  s[t] = v; __syncthreads();
  for (int off = 32; off; off >>= 1) { if (t < off) s[t] = fmaxf(s[t], s[t + off]); __syncthreads(); }
  float mx = s[0]; __syncthreads();
  float e = expf(v - mx);
  s[t] = e; __syncthreads();
  for (int off = 32; off; off >>= 1) { if (t < off) s[t] += s[t + off]; __syncthreads(); }
  dst[row * 64 + t] = e / s[0];
}

// DNC allocation weighting: per-batch sort of usage, exclusive cumprod
__global__ void k_alloc(const float* __restrict__ usage, float* __restrict__ al){
  __shared__ float su[64]; __shared__ int phi[64]; __shared__ float sa[64];
  int b = blockIdx.x, t = threadIdx.x;
  su[t] = EPS_ + (1.0f - EPS_) * usage[b * 64 + t];
  phi[t] = t;
  __syncthreads();
  for (int k = 2; k <= 64; k <<= 1)
    for (int j = k >> 1; j > 0; j >>= 1) {
      int ixj = t ^ j;
      if (ixj > t) {
        bool asc = ((t & k) == 0);
        float a = su[t], c = su[ixj];
        if ((a > c) == asc) {
          su[t] = c; su[ixj] = a;
          int p = phi[t]; phi[t] = phi[ixj]; phi[ixj] = p;
        }
      }
      __syncthreads();
    }
  if (t == 0) {
    float excl = 1.0f;
    for (int i = 0; i < 64; ++i) { float ui = su[i]; sa[i] = (1.0f - ui) * excl; excl *= ui; }
  }
  __syncthreads();
  al[b * 64 + phi[t]] = sa[t];
}

__global__ void k_ww(const float* __restrict__ xia, const float* __restrict__ al,
                     const float* __restrict__ wcw, float* __restrict__ ww){
  int idx = blockIdx.x * blockDim.x + threadIdx.x;
  if (idx >= B_ * M_) return;
  int b = idx >> 6;
  const float* xb = xia + (size_t)b * IFP_;
  float ag = xb[3593], wg = xb[3594];
  ww[idx] = wg * (ag * al[idx] + (1.0f - ag) * wcw[idx]);
}

__global__ void k_memupd(float* __restrict__ mem, const float* __restrict__ ww,
                         const float* __restrict__ xia){
  int idx = blockIdx.x * blockDim.x + threadIdx.x;
  if (idx >= B_ * M_ * D_) return;
  int b = idx >> 15, m = (idx >> 9) & 63, w = idx & 511;
  const float* xb = xia + (size_t)b * IFP_;
  float wwv = ww[b * M_ + m];
  mem[idx] = mem[idx] * (1.0f - wwv * xb[2565 + w]) + wwv * xb[3077 + w];
}

__global__ void k_link(float* __restrict__ link, const float* __restrict__ ww,
                       const float* __restrict__ prec){
  int idx = blockIdx.x * blockDim.x + threadIdx.x;
  if (idx >= B_ * M_ * M_) return;
  int b = idx >> 12, rem = idx & 4095, i = rem >> 6, j = rem & 63;
  float wi = ww[b * M_ + i], wj = ww[b * M_ + j];
  link[idx] = (i == j) ? 0.0f : (1.0f - wi - wj) * link[idx] + wi * prec[b * M_ + j];
}

__global__ void k_prec(float* __restrict__ prec, const float* __restrict__ ww){
  __shared__ float s[64];
  int b = blockIdx.x, t = threadIdx.x;
  float w = ww[b * 64 + t];
  s[t] = w; __syncthreads();
  for (int off = 32; off; off >>= 1) { if (t < off) s[t] += s[t + off]; __syncthreads(); }
  prec[b * 64 + t] = (1.0f - s[0]) * prec[b * 64 + t] + w;
}

__global__ void k_rwupd(const float* __restrict__ link, const float* __restrict__ rw,
                        const float* __restrict__ cw, const float* __restrict__ modes,
                        float* __restrict__ rw2){
  int idx = blockIdx.x * blockDim.x + threadIdx.x;
  if (idx >= B_ * R_ * M_) return;
  int b = idx >> 8, r = (idx >> 6) & 3, n = idx & 63;
  const float* L  = link + (size_t)b * M_ * M_;
  const float* ro = rw + b * (R_*M_) + r * M_;
  float fw = 0.0f, bw = 0.0f;
  for (int m = 0; m < M_; ++m) { float x = ro[m]; fw += L[n * M_ + m] * x; bw += L[m * M_ + n] * x; }
  const float* md = modes + b * 12 + r * 3;
  rw2[idx] = md[0] * bw + md[1] * fw + md[2] * cw[idx];
}

__global__ void k_rv(const float* __restrict__ rw, const float* __restrict__ mem,
                     float* __restrict__ rv){
  int idx = blockIdx.x * blockDim.x + threadIdx.x;
  if (idx >= B_ * R_ * D_) return;
  int b = idx >> 11, r = (idx >> 9) & 3, w = idx & 511;
  const float* ro = rw + b * (R_*M_) + r * M_;
  const float* mp = mem + (size_t)b * M_ * D_ + w;
  float s = 0.0f;
  for (int m = 0; m < M_; ++m) s += ro[m] * mp[(size_t)m * D_];
  rv[idx] = s;
}

__global__ void k_full(const float* __restrict__ out, const float* __restrict__ rv,
                       float* __restrict__ full){
  int idx = blockIdx.x * blockDim.x + threadIdx.x;
  if (idx >= B_ * (D_ + R_ * D_)) return;
  int b = idx / 2560, j = idx - b * 2560;
  full[idx] = (j < D_) ? out[b * D_ + j] : rv[b * (R_*D_) + (j - D_)];
}

// ======================================================================
// host side
// ======================================================================
static inline int cdiv(int a, int b){ return (a + b - 1) / b; }

struct DirParams {
  const float *bih0,*bhh0,*bih1,*bhh1,*bi,*bo;
  const _Float16 *Wih0,*Whh0,*Wih1,*Whh1,*Wi,*Wo;
};

static void gemm(hipStream_t st, const float* A, int lda, const _Float16* W, int ldw,
                 const float* bias, float* C, int ldc,
                 int Mrows, int N, int Npad, int K, int accum){
  int waves  = (Mrows >> 4) * (Npad / 64);    // Npad % 64 == 0
  int blocks = cdiv(waves * 32, 128);
  k_wmma_gemm<0><<<blocks, 128, 0, st>>>(A, lda, W, ldw, bias, C, ldc, Mrows, N, Npad, K, accum);
}

extern "C" void kernel_launch(void* const* d_in, const int* in_sizes, int n_in,
                              void* d_out, int out_size, void* d_ws, size_t ws_size,
                              hipStream_t stream) {
  (void)in_sizes; (void)n_in; (void)out_size; (void)ws_size;

  const float* emb_src = (const float*)d_in[0];
  const float* emb_tgt = (const float*)d_in[1];
  const float* P[24];
  for (int i = 0; i < 24; ++i) P[i] = (const float*)d_in[2 + i];
  const float* fc_W = (const float*)d_in[26];
  const float* fc_b = (const float*)d_in[27];
  const int*   inp  = (const int*)d_in[28];
  const int*   tgtp = (const int*)d_in[29];

  // bump allocator over d_ws (256B aligned)
  char* ws = (char*)d_ws;
  size_t off = 0;
  auto alloc_f = [&](size_t n)->float* {
    float* p = (float*)(ws + off); off += ((n * 4 + 255) / 256) * 256; return p; };
  auto alloc_h = [&](size_t n)->_Float16* {
    _Float16* p = (_Float16*)(ws + off); off += ((n * 2 + 255) / 256) * 256; return p; };

  auto conv = [&](const float* src, int N, int Npad, int ldsrc, int Kd)->_Float16* {
    _Float16* d = alloc_h((size_t)Npad * Kd);
    int total = Npad * Kd;
    k_convert<<<cdiv(total, 256), 256, 0, stream>>>(src, d, N, ldsrc, Kd, total);
    return d;
  };

  // ---- weight conversion to f16 (once per launch); Npad always %64==0 ----
  // enc
  _Float16* eWih0 = conv(P[0],  2048, 2048, 2560, 512);  // only first 512 cols used (xin tail is zeros)
  _Float16* eWhh0 = conv(P[1],  2048, 2048,  512, 512);
  _Float16* eWih1 = conv(P[4],  2048, 2048,  512, 512);
  _Float16* eWhh1 = conv(P[5],  2048, 2048,  512, 512);
  _Float16* eWi   = conv(P[8],  IF_,  IFP64_, 512, 512);
  _Float16* eWo   = conv(P[10],  512,  512, 2560, 2560);
  // dec
  _Float16* dWih0 = conv(P[12], 2048, 2048, 2560, 512);
  _Float16* dWhh0 = conv(P[13], 2048, 2048,  512, 512);
  _Float16* dWih1 = conv(P[16], 2048, 2048,  512, 512);
  _Float16* dWhh1 = conv(P[17], 2048, 2048,  512, 512);
  _Float16* dWi   = conv(P[20], IF_,  IFP64_, 512, 512);
  _Float16* dWo   = conv(P[22],  512,  512, 2560, 2560);
  _Float16* hfc   = conv(fc_W,  V_,   VP_,   512, 512);

  DirParams enc { P[2],  P[3],  P[6],  P[7],  P[9],  P[11], eWih0,eWhh0,eWih1,eWhh1,eWi,eWo };
  DirParams dec { P[14], P[15], P[18], P[19], P[21], P[23], dWih0,dWhh0,dWih1,dWhh1,dWi,dWo };

  // ---- state block (contiguous, zero-initialized) ----
  float* h0     = alloc_f(B_ * D_);
  float* c0     = alloc_f(B_ * D_);
  float* h1     = alloc_f(B_ * D_);
  float* c1     = alloc_f(B_ * D_);
  float* memory = alloc_f((size_t)B_ * M_ * D_);
  float* link   = alloc_f((size_t)B_ * M_ * M_);
  float* prec   = alloc_f(B_ * M_);
  float* rw     = alloc_f(B_ * R_ * M_);
  float* ww     = alloc_f(B_ * M_);
  float* usage  = alloc_f(B_ * M_);
  long state_elems = 4L*B_*D_ + (long)B_*M_*D_ + (long)B_*M_*M_ + B_*M_ + B_*R_*M_ + B_*M_ + B_*M_;

  // ---- scratch ----
  float* rw2    = alloc_f(B_ * R_ * M_);
  float* gbuf   = alloc_f(B_ * 2048);
  float* x_t    = alloc_f(B_ * D_);
  float* outb   = alloc_f(B_ * D_);
  float* xi     = alloc_f((size_t)B_ * IFP_);
  float* xia    = alloc_f((size_t)B_ * IFP_);
  float* modesb = alloc_f(B_ * R_ * 3);
  float* allocb = alloc_f(B_ * M_);
  float* wcw    = alloc_f(B_ * M_);
  float* cw     = alloc_f(B_ * R_ * M_);
  float* mnorm  = alloc_f(B_ * M_);
  float* wknorm = alloc_f(B_);
  float* rknorm = alloc_f(B_ * R_);
  float* rv     = alloc_f(B_ * R_ * D_);
  float* fullb  = alloc_f(B_ * 2560);
  float* ys     = alloc_f((size_t)B_ * T_ * D_);

  k_zero<<<cdiv((int)state_elems, 256), 256, 0, stream>>>(h0, state_elems);

  auto step = [&](const DirParams& dp, const float* emb, const int* ids, int t, bool wr_y){
    // 1) input embedding
    k_embed<<<64, 256, 0, stream>>>(emb, ids, t, 64, x_t);
    // 2) LSTM layer 0  (x W^T + bih + h W^T + bhh)
    gemm(stream, x_t, D_, dp.Wih0, 512, dp.bih0, gbuf, 2048, B_, 2048, 2048, 512, 0);
    gemm(stream, h0,  D_, dp.Whh0, 512, dp.bhh0, gbuf, 2048, B_, 2048, 2048, 512, 1);
    k_lstm<<<64, 256, 0, stream>>>(gbuf, h0, c0);
    // 3) LSTM layer 1
    gemm(stream, h0,  D_, dp.Wih1, 512, dp.bih1, gbuf, 2048, B_, 2048, 2048, 512, 0);
    gemm(stream, h1,  D_, dp.Whh1, 512, dp.bhh1, gbuf, 2048, B_, 2048, 2048, 512, 1);
    k_lstm<<<64, 256, 0, stream>>>(gbuf, h1, c1);
    k_clip<<<64, 256, 0, stream>>>(h1, outb);
    // 4) interface vector
    gemm(stream, outb, D_, dp.Wi, 512, dp.bi, xi, IFP_, B_, IF_, IFP64_, 512, 0);
    k_xiact<<<cdiv(B_ * IF_, 256), 256, 0, stream>>>(xi, xia);
    k_modes<<<1, 128, 0, stream>>>(xia, modesb);
    // 5) memory step
    k_usage<<<8, 256, 0, stream>>>(xia, ww, rw, usage);
    k_norm512<<<256, 256, 0, stream>>>(memory, mnorm, B_*M_, M_, (long)M_*D_, D_, 0);
    k_norm512<<<4,   256, 0, stream>>>(xia, wknorm, B_, 1, IFP_, 0, 2052);
    k_wcw<<<256, 256, 0, stream>>>(memory, xia, mnorm, wknorm, wcw);
    k_softmax64<<<B_, 64, 0, stream>>>(wcw, wcw);
    k_alloc<<<B_, 64, 0, stream>>>(usage, allocb);
    k_ww<<<8, 256, 0, stream>>>(xia, allocb, wcw, ww);
    k_memupd<<<4096, 256, 0, stream>>>(memory, ww, xia);
    k_link<<<512, 256, 0, stream>>>(link, ww, prec);
    k_prec<<<B_, 64, 0, stream>>>(prec, ww);
    k_norm512<<<256, 256, 0, stream>>>(memory, mnorm, B_*M_, M_, (long)M_*D_, D_, 0);
    k_norm512<<<16,  256, 0, stream>>>(xia, rknorm, B_*R_, R_, IFP_, D_, 0);
    k_cw<<<1024, 256, 0, stream>>>(memory, xia, mnorm, rknorm, cw);
    k_softmax64<<<B_*R_, 64, 0, stream>>>(cw, cw);
    k_rwupd<<<32, 256, 0, stream>>>(link, rw, cw, modesb, rw2);
    k_copy<<<32, 256, 0, stream>>>(rw, rw2, B_*R_*M_);
    k_rv<<<256, 256, 0, stream>>>(rw, memory, rv);
    // 6) decoder output
    if (wr_y) {
      k_full<<<320, 256, 0, stream>>>(outb, rv, fullb);
      gemm(stream, fullb, 2560, dp.Wo, 2560, dp.bo, ys + (size_t)t * D_, T_ * D_,
           B_, D_, D_, 2560, 0);
    }
  };

  for (int t = 0; t < S_; ++t) step(enc, emb_src, inp,  t, false);
  // reset h/c for decoder; memory state carries over
  k_zero<<<cdiv(4 * B_ * D_, 256), 256, 0, stream>>>(h0, 4L * B_ * D_);
  for (int t = 0; t < T_; ++t) step(dec, emb_tgt, tgtp, t, true);

  // final vocab projection: logits[b,t,v] stored transposed as out[b,v,t]
  {
    int Mrows = B_ * T_;                        // 2048
    int waves  = (Mrows >> 4) * (VP_ / 64);     // 128 * 157
    int blocks = cdiv(waves * 32, 128);
    k_wmma_gemm<1><<<blocks, 128, 0, stream>>>(ys, D_, hfc, 512, fc_b,
                                               (float*)d_out, 0, Mrows, V_, VP_, 512, 0);
  }
}
